// TwoLayerGAT_47691316854882
// MI455X (gfx1250) — compile-verified, matrix operand order checked by reading, not run on hardware
//
#include <hip/hip_runtime.h>
#include <hip/hip_bf16.h>

typedef __attribute__((ext_vector_type(16))) _Float16 v16h;
typedef __attribute__((ext_vector_type(8)))  _Float16 v8h;
typedef __attribute__((ext_vector_type(8)))  float    v8f;

union F16Frag { v16h v; v8h h[2]; };

#define NEG_SLOPE 0.2f
#define TPB 256

// ---------- helpers ----------
__device__ __forceinline__ unsigned f2ord(float f) {
  unsigned u = __float_as_uint(f);
  return (u & 0x80000000u) ? ~u : (u | 0x80000000u);
}
__device__ __forceinline__ float ord2f(unsigned u) {
  unsigned v = (u & 0x80000000u) ? (u & 0x7fffffffu) : ~u;
  return __uint_as_float(v);
}
// edge e in [0,E) -> graph edge; e in [E,E+N) -> self loop (e-E, e-E)
__device__ __forceinline__ void edge_sd(const int* __restrict__ ei, int E, int e,
                                        int& s, int& d) {
  if (e < E) { s = ei[e]; d = ei[E + e]; } else { s = e - E; d = e - E; }
}

// ---------- utility kernels ----------
__global__ void k_fill_f32(float* __restrict__ p, float v, long long n) {
  long long t = (long long)blockIdx.x * blockDim.x + threadIdx.x;
  if (t < n) p[t] = v;
}
__global__ void k_fill_u32(unsigned* __restrict__ p, unsigned v, long long n) {
  long long t = (long long)blockIdx.x * blockDim.x + threadIdx.x;
  if (t < n) p[t] = v;
}
__global__ void k_cvt_f16(const float* __restrict__ s, _Float16* __restrict__ d,
                          long long n) {
  long long t = (long long)blockIdx.x * blockDim.x + threadIdx.x;
  if (t < n) d[t] = (_Float16)s[t];
}
// W [K,Nc] f32 row-major -> Wt [Nc,K] f16 (so B columns are contiguous)
__global__ void k_cvt_transpose_f16(const float* __restrict__ W,
                                    _Float16* __restrict__ Wt, int K, int Nc) {
  int t = blockIdx.x * blockDim.x + threadIdx.x;
  if (t >= K * Nc) return;
  int k = t / Nc, c = t - k * Nc;
  Wt[(size_t)c * K + k] = (_Float16)W[t];
}

// ---------- WMMA GEMM: C[M,Nc] = A[M,128] * B[128,Nc], Bt = B transposed ----------
// NTN = Nc/16 (compile-time so the epilogue uses immediate-offset stores).
template <int NTN>
__global__ void k_wmma_gemm(const _Float16* __restrict__ A,
                            const _Float16* __restrict__ Bt,
                            float* __restrict__ C,
                            int M, int ntiles) {
  constexpr int K  = 128;
  constexpr int Nc = NTN * 16;
  int wave = (int)((blockIdx.x * (unsigned)blockDim.x + threadIdx.x) >> 5);
  if (wave >= ntiles) return;                 // wave-uniform; EXEC stays all-ones
  int lane = threadIdx.x & 31;
  int mt = wave / NTN, nt = wave - mt * NTN;
  int l16 = lane & 15;
  int khi = (lane >> 4) << 3;                 // 0 for lanes 0-15, 8 for lanes 16-31

  int arow = mt * 16 + l16; if (arow >= M) arow = M - 1;   // clamp, keep EXEC full
  const _Float16* ap = A  + (size_t)arow * K;
  const _Float16* bp = Bt + (size_t)(nt * 16 + l16) * K;

  v8f acc = {};
#pragma unroll
  for (int k0 = 0; k0 < K; k0 += 32) {
    F16Frag a, b;
    a.h[0] = *(const v8h*)(ap + k0 + khi);        // K = k0 + {0..7 | 8..15}
    a.h[1] = *(const v8h*)(ap + k0 + 16 + khi);   // K = k0 + {16..23 | 24..31}
    b.h[0] = *(const v8h*)(bp + k0 + khi);
    b.h[1] = *(const v8h*)(bp + k0 + 16 + khi);
    acc = __builtin_amdgcn_wmma_f32_16x16x32_f16(
        false, a.v, false, b.v, (short)0, acc, false, false);
  }

  int rhi = (lane >> 4) << 3;                 // lanes 16-31 hold rows M=8..15
  int r0  = mt * 16 + rhi;
  float* cp = C + (size_t)r0 * Nc + nt * 16 + l16;
  if (mt * 16 + 16 <= M) {                    // full tile: immediate-offset stores
#pragma unroll
    for (int r = 0; r < 8; ++r) cp[r * Nc] = acc[r];
  } else {
#pragma unroll
    for (int r = 0; r < 8; ++r)
      if (r0 + r < M) cp[r * Nc] = acc[r];
  }
}

// ---------- per-node attention scores ----------
template <int H, int C>
__global__ void k_scores(const float* __restrict__ h, const float* __restrict__ a_src,
                         const float* __restrict__ a_dst, float* __restrict__ s,
                         float* __restrict__ d, int N) {
  int t = blockIdx.x * blockDim.x + threadIdx.x;
  if (t >= N * H) return;
  int n = t / H, hh = t % H;
  const float* hp = h + (size_t)n * H * C + hh * C;
  const float* as = a_src + hh * C;
  const float* ad = a_dst + hh * C;
  float ss = 0.f, dd = 0.f;
#pragma unroll 8
  for (int c = 0; c < C; ++c) { float v = hp[c]; ss += v * as[c]; dd += v * ad[c]; }
  s[t] = ss; d[t] = dd;
}

// ---------- edge pass 1: segment max of leaky-relu logits ----------
template <int H>
__global__ void k_edge_max(const int* __restrict__ ei, int E, int N,
                           const float* __restrict__ s, const float* __restrict__ d,
                           unsigned* __restrict__ m) {
  int t = blockIdx.x * blockDim.x + threadIdx.x;
  int total = (E + N) * H;
  if (t >= total) return;
  int e = t / H, hh = t % H;
  int src, dst; edge_sd(ei, E, e, src, dst);
  float v = s[(size_t)src * H + hh] + d[(size_t)dst * H + hh];
  v = (v > 0.f) ? v : NEG_SLOPE * v;
  atomicMax(m + (size_t)dst * H + hh, f2ord(v));
}

// ---------- edge pass 2: ex = exp(logit - max), segment-sum denominators ----------
template <int H>
__global__ void k_edge_exp(const int* __restrict__ ei, int E, int N,
                           const float* __restrict__ s, const float* __restrict__ d,
                           const unsigned* __restrict__ m, float* __restrict__ ex,
                           float* __restrict__ den) {
  int t = blockIdx.x * blockDim.x + threadIdx.x;
  int total = (E + N) * H;
  if (t >= total) return;
  int e = t / H, hh = t % H;
  int src, dst; edge_sd(ei, E, e, src, dst);
  float v = s[(size_t)src * H + hh] + d[(size_t)dst * H + hh];
  v = (v > 0.f) ? v : NEG_SLOPE * v;
  float ev = __expf(v - ord2f(m[(size_t)dst * H + hh]));
  ex[t] = ev;
  atomicAdd(den + (size_t)dst * H + hh, ev);
}

// ---------- edge pass 3: out[dst] += alpha * feat[src]  (float4 + f32 L2 atomics) ----------
template <int H, int C>
__global__ void k_edge_aggr(const int* __restrict__ ei, int E, int N,
                            const float* __restrict__ feat, const float* __restrict__ ex,
                            const float* __restrict__ den, float* __restrict__ out) {
  constexpr int CQ = C / 4;                   // float4 groups per head
  long long t = (long long)blockIdx.x * blockDim.x + threadIdx.x;
  long long total = (long long)(E + N) * H * CQ;
  if (t >= total) return;
  int q  = (int)(t % CQ);
  int eh = (int)(t / CQ);
  int hh = eh % H;
  int e  = eh / H;
  int src, dst; edge_sd(ei, E, e, src, dst);
  float alpha = ex[(size_t)e * H + hh] / den[(size_t)dst * H + hh];
  int f = hh * C + q * 4;
  const float4 fv = *(const float4*)(feat + (size_t)src * (H * C) + f);
  float* op = out + (size_t)dst * (H * C) + f;
  atomicAdd(op + 0, fv.x * alpha);
  atomicAdd(op + 1, fv.y * alpha);
  atomicAdd(op + 2, fv.z * alpha);
  atomicAdd(op + 3, fv.w * alpha);
}

// ---------- bias + relu + convert to f16 (feeds GEMM2) ----------
template <int F>
__global__ void k_bias_relu_cvt(const float* __restrict__ in, const float* __restrict__ bias,
                                _Float16* __restrict__ out, long long n) {
  long long t = (long long)blockIdx.x * blockDim.x + threadIdx.x;
  if (t >= n) return;
  float v = in[t] + bias[(int)(t % F)];
  out[t] = (_Float16)(v > 0.f ? v : 0.f);
}

// ---------- final bias add in-place on d_out ----------
template <int F>
__global__ void k_bias_add(float* __restrict__ out, const float* __restrict__ bias,
                           long long n) {
  long long t = (long long)blockIdx.x * blockDim.x + threadIdx.x;
  if (t < n) out[t] += bias[(int)(t % F)];
}

// ---------- host ----------
static inline size_t align256(size_t x) { return (x + 255) & ~(size_t)255; }
static inline unsigned nblk(long long n) { return (unsigned)((n + TPB - 1) / TPB); }

extern "C" void kernel_launch(void* const* d_in, const int* in_sizes, int n_in,
                              void* d_out, int out_size, void* d_ws, size_t ws_size,
                              hipStream_t stream) {
  (void)n_in; (void)out_size; (void)ws_size;
  const float* x   = (const float*)d_in[0];
  const int*   ei  = (const int*)d_in[1];   // [2,E] int32: row 0 = src, row 1 = dst
  const float* W1  = (const float*)d_in[3];
  const float* a1s = (const float*)d_in[4];
  const float* a1d = (const float*)d_in[5];
  const float* b1  = (const float*)d_in[6];
  const float* W2  = (const float*)d_in[7];
  const float* a2s = (const float*)d_in[8];
  const float* a2d = (const float*)d_in[9];
  const float* b2  = (const float*)d_in[10];
  float* out = (float*)d_out;

  constexpr int F_in = 128, H1 = 4, C1 = 32, F1 = 128, F2 = 64;
  const int N  = in_sizes[0] / F_in;   // 50000
  const int E  = in_sizes[1] / 2;      // 800000
  const int Ep = E + N;                // + self loops

  // workspace bump allocator
  char* base = (char*)d_ws; size_t off = 0;
#define ALLOC(ty, name, count) ty* name = (ty*)(base + off); off = align256(off + (size_t)(count) * sizeof(ty))
  ALLOC(_Float16, x16,   (size_t)N * F_in);
  ALLOC(_Float16, W1t,   (size_t)F_in * F1);
  ALLOC(_Float16, W2t,   (size_t)F1 * F2);
  ALLOC(float,    h1,    (size_t)N * F1);
  ALLOC(float,    s1,    (size_t)N * H1);
  ALLOC(float,    d1,    (size_t)N * H1);
  ALLOC(unsigned, m1,    (size_t)N * H1);
  ALLOC(float,    den1,  (size_t)N * H1);
  ALLOC(float,    ex1,   (size_t)Ep * H1);
  ALLOC(float,    att1,  (size_t)N * F1);
  ALLOC(_Float16, h2in,  (size_t)N * F1);
  ALLOC(float,    h2,    (size_t)N * F2);
  ALLOC(float,    s2,    (size_t)N);
  ALLOC(float,    d2,    (size_t)N);
  ALLOC(unsigned, m2,    (size_t)N);
  ALLOC(float,    den2,  (size_t)N);
  ALLOC(float,    ex2,   (size_t)Ep);
#undef ALLOC

  // ---- init accumulators every call (harness does not re-poison) ----
  k_fill_u32<<<nblk((long long)N*H1), TPB, 0, stream>>>(m1, 0u, (long long)N*H1);
  k_fill_f32<<<nblk((long long)N*H1), TPB, 0, stream>>>(den1, 0.f, (long long)N*H1);
  k_fill_f32<<<nblk((long long)N*F1), TPB, 0, stream>>>(att1, 0.f, (long long)N*F1);
  k_fill_u32<<<nblk(N), TPB, 0, stream>>>(m2, 0u, N);
  k_fill_f32<<<nblk(N), TPB, 0, stream>>>(den2, 0.f, N);
  k_fill_f32<<<nblk((long long)N*F2), TPB, 0, stream>>>(out, 0.f, (long long)N*F2);

  // ---- precision prep: f16 operands for the WMMA GEMMs ----
  k_cvt_f16<<<nblk((long long)N*F_in), TPB, 0, stream>>>(x, x16, (long long)N*F_in);
  k_cvt_transpose_f16<<<nblk(F_in*F1), TPB, 0, stream>>>(W1, W1t, F_in, F1);
  k_cvt_transpose_f16<<<nblk(F1*F2),  TPB, 0, stream>>>(W2, W2t, F1, F2);

  // ---- layer 1: GEMM (WMMA), scores, softmax, aggregate ----
  {
    constexpr int NTN = F1 / 16;
    int ntiles = ((N + 15) / 16) * NTN;
    k_wmma_gemm<NTN><<<nblk((long long)ntiles * 32), TPB, 0, stream>>>(x16, W1t, h1, N, ntiles);
  }
  k_scores<H1, C1><<<nblk((long long)N*H1), TPB, 0, stream>>>(h1, a1s, a1d, s1, d1, N);
  k_edge_max<H1><<<nblk((long long)Ep*H1), TPB, 0, stream>>>(ei, E, N, s1, d1, m1);
  k_edge_exp<H1><<<nblk((long long)Ep*H1), TPB, 0, stream>>>(ei, E, N, s1, d1, m1, ex1, den1);
  k_edge_aggr<H1, C1><<<nblk((long long)Ep*H1*(C1/4)), TPB, 0, stream>>>(ei, E, N, h1, ex1, den1, att1);
  k_bias_relu_cvt<F1><<<nblk((long long)N*F1), TPB, 0, stream>>>(att1, b1, h2in, (long long)N*F1);

  // ---- layer 2: GEMM (WMMA), scores (H=1,C=64), softmax, aggregate into d_out ----
  {
    constexpr int NTN = F2 / 16;
    int ntiles = ((N + 15) / 16) * NTN;
    k_wmma_gemm<NTN><<<nblk((long long)ntiles * 32), TPB, 0, stream>>>(h2in, W2t, h2, N, ntiles);
  }
  k_scores<1, F2><<<nblk(N), TPB, 0, stream>>>(h2, a2s, a2d, s2, d2, N);
  k_edge_max<1><<<nblk(Ep), TPB, 0, stream>>>(ei, E, N, s2, d2, m2);
  k_edge_exp<1><<<nblk(Ep), TPB, 0, stream>>>(ei, E, N, s2, d2, m2, ex2, den2);
  k_edge_aggr<1, F2><<<nblk((long long)Ep*(F2/4)), TPB, 0, stream>>>(ei, E, N, h2, ex2, den2, out);
  k_bias_add<F2><<<nblk((long long)N*F2), TPB, 0, stream>>>(out, b2, (long long)N*F2);
}
